// MolecularEncoderExtended_42442866819262
// MI455X (gfx1250) — compile-verified, matrix operand order checked by reading, not run on hardware
//
#include <hip/hip_runtime.h>
#include <hip/hip_bf16.h>

typedef _Float16 v16h __attribute__((ext_vector_type(16)));
typedef float    v8f  __attribute__((ext_vector_type(8)));
typedef float    v2f  __attribute__((ext_vector_type(2)));

#define BATCH 8
#define NATOM 512
#define KGBF  64
#define DIM   128
#define NTYPES 66
#define CUTOFF_F 5.0f

#if __has_builtin(__builtin_amdgcn_wmma_f32_16x16x4_f32)
#define HAVE_WMMA_F32 1
#else
#define HAVE_WMMA_F32 0
#endif

// ---------------------------------------------------------------------------
// Kernel 1: per-batch histogram of atom types over non-pad atoms (t>0).
// ---------------------------------------------------------------------------
__global__ void hist_kernel(const int* __restrict__ atoms, float* __restrict__ cnt) {
    __shared__ int h[NTYPES];
    const int b = blockIdx.x, tid = threadIdx.x;
    if (tid < NTYPES) h[tid] = 0;
    __syncthreads();
    for (int j = tid; j < NATOM; j += 128) {
        int t = atoms[b * NATOM + j];
        if (t > 0) atomicAdd(&h[t], 1);
    }
    __syncthreads();
    if (tid < NTYPES) cnt[b * NTYPES + tid] = (float)h[tid];
}

// ---------------------------------------------------------------------------
// Kernel 2: atom_3d[b,i,k] = sum_j [d_ij<=5 & !pad_j] exp(-((d_ij-c_k)/w_k)^2)
//                           + sum_t cnt[b][t] * type_bias[a_i*66+t, k]
// exp(-x^2) folded to a single v_exp_f32: exp2(-(d*s - c_k*s)^2), s=sqrt(log2e)/w.
// ---------------------------------------------------------------------------
__global__ void gauss_sum_kernel(const int*   __restrict__ atoms,
                                 const float* __restrict__ pos,
                                 const float* __restrict__ centers,
                                 const float* __restrict__ widths,
                                 const float* __restrict__ type_bias,
                                 const float* __restrict__ cnt,
                                 float*       __restrict__ a3) {
    __shared__ __align__(16) float sdist[NATOM];
    __shared__ float red[256];

    const int i = blockIdx.x, b = blockIdx.y;
    const int tid = threadIdx.x;
    const int base = b * NATOM;

    const float px = pos[(base + i) * 3 + 0];
    const float py = pos[(base + i) * 3 + 1];
    const float pz = pos[(base + i) * 3 + 2];

    for (int j = tid; j < NATOM; j += 256) {
        float dx = px - pos[(base + j) * 3 + 0];
        float dy = py - pos[(base + j) * 3 + 1];
        float dz = pz - pos[(base + j) * 3 + 2];
        float sq = dx * dx + dy * dy + dz * dz;
        float d  = __builtin_amdgcn_sqrtf(sq);          // sqrt(0)=0 matches ref
        if (atoms[base + j] == 0) d = 1e30f;            // pad_j -> fails cutoff
        sdist[j] = d;
    }
    __syncthreads();

    const int k      = tid & 63;
    const int stripe = tid >> 6;                        // wave-uniform
    const float s    = 1.2011224087864498f / widths[k]; // sqrt(log2 e)/w
    const float cks  = centers[k] * s;

    float acc = 0.0f;
    const float4* dptr = (const float4*)&sdist[stripe * 128];
    #pragma unroll 4
    for (int q = 0; q < 32; ++q) {
        float4 d4 = dptr[q];
        float dmin = fminf(fminf(d4.x, d4.y), fminf(d4.z, d4.w));
        if (dmin > CUTOFF_F) continue;                  // wave-coherent skip
        if (d4.x <= CUTOFF_F) { float a = __builtin_fmaf(d4.x, s, -cks); acc += __builtin_amdgcn_exp2f(-a * a); }
        if (d4.y <= CUTOFF_F) { float a = __builtin_fmaf(d4.y, s, -cks); acc += __builtin_amdgcn_exp2f(-a * a); }
        if (d4.z <= CUTOFF_F) { float a = __builtin_fmaf(d4.z, s, -cks); acc += __builtin_amdgcn_exp2f(-a * a); }
        if (d4.w <= CUTOFF_F) { float a = __builtin_fmaf(d4.w, s, -cks); acc += __builtin_amdgcn_exp2f(-a * a); }
    }
    red[tid] = acc;
    __syncthreads();

    if (stripe == 0) {
        float tot = red[k] + red[k + 64] + red[k + 128] + red[k + 192];
        const int ai = atoms[base + i];
        const float* tb = type_bias + (size_t)(ai * NTYPES) * KGBF + k;
        const float* cb = cnt + b * NTYPES;
        float bsum = 0.0f;
        #pragma unroll 5
        for (int t = 1; t < NTYPES; ++t) bsum = __builtin_fmaf(cb[t], tb[t * KGBF], bsum);
        a3[(size_t)(base + i) * KGBF + k] = tot + bsum;
    }
}

// ---------------------------------------------------------------------------
// Async bulk copy global -> LDS, 16B per lane per issue (ASYNCcnt-tracked).
// ldsAddr: LDS byte address (low 32 bits of flat shared address).
// ---------------------------------------------------------------------------
__device__ __forceinline__ void async_copy_b128(unsigned int ldsAddr, const void* gaddr) {
    unsigned long long ga = (unsigned long long)(uintptr_t)gaddr;
    asm volatile("global_load_async_to_lds_b128 %0, %1, off"
                 :: "v"(ldsAddr), "v"(ga) : "memory");
}
__device__ __forceinline__ void async_wait0() {
    asm volatile("s_wait_asynccnt 0x0" ::: "memory");
}

// ---------------------------------------------------------------------------
// Kernel 3: proj = atom_3d @ W^T via WMMA, + b + atom_embed[atoms],
// LayerNorm over D=128, pad-mask. One block per 16-row tile; 8 waves each
// own a 16-col tile -> full D=128. W/A staged to LDS with async-to-LDS.
// ---------------------------------------------------------------------------
__global__ void proj_ln_kernel(const float* __restrict__ a3,         // [4096,64]
                               const int*   __restrict__ atoms,      // [4096]
                               const float* __restrict__ atom_embed, // [66,128]
                               const float* __restrict__ W,          // [128,64]
                               const float* __restrict__ bvec,       // [128]
                               const float* __restrict__ gamma,
                               const float* __restrict__ beta,
                               float*       __restrict__ out) {      // [4096,128]
    __shared__ __align__(16) float Wlds[DIM * KGBF];   // 32 KB
    __shared__ __align__(16) float Alds[16 * KGBF];    //  4 KB
    __shared__ float fused[16][DIM];                   //  8 KB
    __shared__ float psum[16][16], psq[16][16];
    __shared__ float smu[16], srs[16];
    __shared__ int   sAtom[16];

    const int tid  = threadIdx.x;
    const int row0 = blockIdx.x * 16;

    // ---- async stage: W (32KB = 8 issues x 256 lanes x 16B), A rows (4KB) ----
    {
        const unsigned int wlds0 = (unsigned int)(uintptr_t)&Wlds[0];
        const unsigned int alds0 = (unsigned int)(uintptr_t)&Alds[0];
        #pragma unroll
        for (int it = 0; it < 8; ++it) {
            const unsigned int off = (unsigned int)(it * 256 + tid) * 16u;
            async_copy_b128(wlds0 + off, (const char*)W + off);
        }
        const unsigned int aoff = (unsigned int)tid * 16u;
        async_copy_b128(alds0 + aoff, (const char*)(a3 + (size_t)row0 * KGBF) + aoff);
    }
    if (tid < 16) sAtom[tid] = atoms[row0 + tid];
    async_wait0();
    __syncthreads();

    const int wave  = tid >> 5;     // n-tile: cols [wave*16, wave*16+16)
    const int lane  = tid & 31;
    const int l16   = lane & 15;
    const int khalf = lane >> 4;

    v8f acc = {};
    const int arow = l16;                 // A row within tile (M = lane%16)
    const int bcol = wave * 16 + l16;     // global output column (N = lane%16)

#if HAVE_WMMA_F32
    // V_WMMA_F32_16X16X4_F32: A 16x4 (lane m=lane%16, VGPR0/1 = K 2*(lane/16)+{0,1});
    // B 4x16 mirrored. K=64 -> 16 chained WMMAs, full f32 precision.
    {
        const float* pa = &Alds[arow * KGBF + 2 * khalf];
        const float* pb = &Wlds[bcol * KGBF + 2 * khalf];
        #pragma unroll
        for (int kk = 0; kk < KGBF; kk += 4) {
            v2f a, bb;
            a[0]  = pa[kk]; a[1]  = pa[kk + 1];
            bb[0] = pb[kk]; bb[1] = pb[kk + 1];
            acc = __builtin_amdgcn_wmma_f32_16x16x4_f32(false, a, false, bb,
                                                        (short)0, acc, false, false);
        }
    }
#else
    // Fallback: codegen-confirmed f16 WMMA (K=64 -> 2 chained 16x16x32).
    #pragma unroll
    for (int kk = 0; kk < KGBF; kk += 32) {
        v16h af, bf;
        const float* pa = &Alds[arow * KGBF + kk + 8 * khalf];
        const float* pb = &Wlds[bcol * KGBF + kk + 8 * khalf];
        #pragma unroll
        for (int h = 0; h < 8; ++h) { af[h] = (_Float16)pa[h];          bf[h] = (_Float16)pb[h]; }
        #pragma unroll
        for (int h = 0; h < 8; ++h) { af[8 + h] = (_Float16)pa[16 + h]; bf[8 + h] = (_Float16)pb[16 + h]; }
        acc = __builtin_amdgcn_wmma_f32_16x16x32_f16(false, af, false, bf,
                                                     (short)0, acc, false, false);
    }
#endif

    // D layout: VGPR v, lanes0-15 -> (M=v, N=lane); lanes16-31 -> (M=v+8, N=lane-16)
    #pragma unroll
    for (int v = 0; v < 8; ++v) {
        int m   = v + 8 * khalf;
        int col = wave * 16 + l16;
        fused[m][col] = acc[v] + bvec[col] + atom_embed[sAtom[m] * DIM + col];
    }
    __syncthreads();

    // LayerNorm over D=128: 16 threads per row, 8 elems per thread
    {
        int r = tid >> 4, c = tid & 15;
        float s = 0.f, s2 = 0.f;
        #pragma unroll
        for (int e = 0; e < 8; ++e) { float x = fused[r][c * 8 + e]; s += x; s2 += x * x; }
        psum[r][c] = s; psq[r][c] = s2;
    }
    __syncthreads();
    if ((tid & 15) == 0) {
        int r = tid >> 4;
        float s = 0.f, s2 = 0.f;
        #pragma unroll
        for (int c = 0; c < 16; ++c) { s += psum[r][c]; s2 += psq[r][c]; }
        float mu  = s * (1.0f / DIM);
        float var = s2 * (1.0f / DIM) - mu * mu;
        smu[r] = mu;
        srs[r] = __builtin_amdgcn_rsqf(var + 1e-5f);
    }
    __syncthreads();
    {
        int r = tid >> 4, c = tid & 15;
        int grow = row0 + r;
        bool pad = (sAtom[r] == 0);
        float mu = smu[r], rs = srs[r];
        #pragma unroll
        for (int e = 0; e < 8; ++e) {
            int d = c * 8 + e;
            float x = (fused[r][d] - mu) * rs * gamma[d] + beta[d];
            out[(size_t)grow * DIM + d] = pad ? 0.0f : x;
        }
    }
}

// ---------------------------------------------------------------------------
extern "C" void kernel_launch(void* const* d_in, const int* in_sizes, int n_in,
                              void* d_out, int out_size, void* d_ws, size_t ws_size,
                              hipStream_t stream) {
    const int*   atoms      = (const int*)  d_in[0];
    const float* pos        = (const float*)d_in[1];
    const float* centers    = (const float*)d_in[2];
    const float* widths     = (const float*)d_in[3];
    const float* type_bias  = (const float*)d_in[4];
    const float* atom_embed = (const float*)d_in[5];
    const float* W          = (const float*)d_in[6];
    const float* bvec       = (const float*)d_in[7];
    const float* gamma      = (const float*)d_in[8];
    const float* beta       = (const float*)d_in[9];
    float* out = (float*)d_out;

    float* a3  = (float*)d_ws;                       // 4096*64 floats = 1 MB
    float* cnt = a3 + (size_t)BATCH * NATOM * KGBF;  // 8*66 floats

    hist_kernel<<<dim3(BATCH), dim3(128), 0, stream>>>(atoms, cnt);
    gauss_sum_kernel<<<dim3(NATOM, BATCH), dim3(256), 0, stream>>>(
        atoms, pos, centers, widths, type_bias, cnt, a3);
    proj_ln_kernel<<<dim3((BATCH * NATOM) / 16), dim3(256), 0, stream>>>(
        a3, atoms, atom_embed, W, bvec, gamma, beta, out);
}
